// BidirectionalLSTMComplex_69698729280013
// MI455X (gfx1250) — compile-verified, matrix operand order checked by reading, not run on hardware
//
#include <hip/hip_runtime.h>
#include <hip/hip_bf16.h>

typedef __attribute__((ext_vector_type(16))) _Float16 v16h;
typedef __attribute__((ext_vector_type(8)))  _Float16 v8h;
typedef __attribute__((ext_vector_type(8)))  float    v8f;
typedef __attribute__((ext_vector_type(4)))  float    v4f;

#define SEQ    1024
#define BATCH  2048
#define HID    32
#define NGATE  128   // 4*H
#define MTILE  16    // batch rows per wave (WMMA M)

__device__ __forceinline__ float fexp(float x) { return __builtin_amdgcn_exp2f(x * 1.4426950408889634f); }
__device__ __forceinline__ float frcp(float x) { return __builtin_amdgcn_rcpf(x); }
__device__ __forceinline__ float fsig(float x) { return frcp(1.0f + fexp(-x)); }
__device__ __forceinline__ float ftanh(float x) { float e = fexp(-2.0f * x); return (1.0f - e) * frcp(1.0f + e); }

// One wave (32 threads) per 16-row batch tile; 128 independent blocks.
__global__ __launch_bounds__(32) void bilstm_wmma_kernel(
    const float* __restrict__ x,
    const float* __restrict__ wih_f0, const float* __restrict__ whh_f0, const float* __restrict__ b_f0,
    const float* __restrict__ wih_f1, const float* __restrict__ whh_f1, const float* __restrict__ b_f1,
    const float* __restrict__ wih_b0, const float* __restrict__ whh_b0, const float* __restrict__ b_b0,
    const float* __restrict__ wih_b1, const float* __restrict__ whh_b1, const float* __restrict__ b_b1,
    const float* __restrict__ wlin, const float* __restrict__ blin,
    float* __restrict__ out)
{
    // B operands pre-laid-out for v_wmma_f32_16x16x32_f16:
    // lane L holds col n'=L&15, K = kb..kb+7 (half0) and kb+16..kb+23 (half1), kb=8*(L>>4).
    __shared__ __align__(16) _Float16 Bh[6][8][2][32][8];   // [mat][Ntile][half][lane][8] = 48KB
    __shared__ __align__(16) _Float16 hRow[4][MTILE][HID];  // f0,f1,b0,b1 row-major f16 (A src + out src)
    __shared__ float sb[4][NGATE];      // biases
    __shared__ float swcol[2][NGATE];   // Wih_f0, Wih_b0 columns (IN=1 -> rank-1 term)
    __shared__ float swlin[2 * HID];
    __shared__ float sblin;
    __shared__ float sx[2][MTILE];      // x_t, xb_t for this batch tile

    const int lane = threadIdx.x;
    const int np   = lane & 15;
    const int kb   = (lane >> 4) * 8;
    const int b0r  = blockIdx.x * MTILE;

    { // ---- one-time staging: weights -> LDS f16 in B-operand layout ----
        const float* mats[6] = { whh_f0, wih_f1, whh_f1, whh_b0, wih_b1, whh_b1 };
#pragma unroll
        for (int m = 0; m < 6; ++m) {
            const float* W = mats[m];
#pragma unroll
            for (int t = 0; t < 8; ++t) {
                const float* rp = W + (t * 16 + np) * HID + kb;  // row = gate index 16t+n'
                v4f a0 = *(const v4f*)(rp);
                v4f a1 = *(const v4f*)(rp + 4);
                v4f c0 = *(const v4f*)(rp + 16);
                v4f c1 = *(const v4f*)(rp + 20);
#pragma unroll
                for (int j = 0; j < 4; ++j) {
                    Bh[m][t][0][lane][j]     = (_Float16)a0[j];
                    Bh[m][t][0][lane][4 + j] = (_Float16)a1[j];
                    Bh[m][t][1][lane][j]     = (_Float16)c0[j];
                    Bh[m][t][1][lane][4 + j] = (_Float16)c1[j];
                }
            }
        }
        for (int i = lane; i < NGATE; i += 32) {
            sb[0][i] = b_f0[i]; sb[1][i] = b_f1[i]; sb[2][i] = b_b0[i]; sb[3][i] = b_b1[i];
            swcol[0][i] = wih_f0[i]; swcol[1][i] = wih_b0[i];
        }
        for (int i = lane; i < 2 * HID; i += 32) swlin[i] = wlin[i];
        if (lane == 0) sblin = blin[0];
        v8h z = {};
        for (int i = lane; i < 4 * MTILE * HID / 8; i += 32) ((v8h*)&hRow[0][0][0])[i] = z;
    }
    __syncthreads();

    // c-states live in VGPRs, C/D layout: [tilepair*8 + r], row m = 8*(lane>>4)+r, col = tp*16 + np
    float c_f0[16], c_f1[16], c_b0[16], c_b1[16];
#pragma unroll
    for (int i = 0; i < 16; ++i) { c_f0[i] = 0.f; c_f1[i] = 0.f; c_b0[i] = 0.f; c_b1[i] = 0.f; }

    // A operand from row-major h tile: row = np, 16B chunks at kb*2 and (kb+16)*2 (16B aligned).
    auto loadA = [&](int cellIdx) -> v16h {
        union { v16h v; v8h h[2]; } u;
        u.h[0] = *(const v8h*)&hRow[cellIdx][np][kb];
        u.h[1] = *(const v8h*)&hRow[cellIdx][np][kb + 16];
        return u.v;
    };
    auto loadB = [&](int mat, int t) -> v16h {
        union { v16h v; v8h h[2]; } u;
        u.h[0] = *(const v8h*)&Bh[mat][t][0][lane][0];
        u.h[1] = *(const v8h*)&Bh[mat][t][1][lane][0];
        return u.v;
    };

    // One LSTM cell for this 16-row tile. Gate tiles: i=t, f=t+2, g=t+4, o=t+6 per hidden-col pair tp.
    auto cell = [&](int biasIdx, int xIdx, v16h A1, int mat1, bool dual, v16h A2, int mat2,
                    float* cst, int hDst) {
        float xr[8];
        if (xIdx >= 0) {
#pragma unroll
            for (int r = 0; r < 8; ++r) xr[r] = sx[xIdx][(lane >> 4) * 8 + r];
        }
#pragma unroll
        for (int tp = 0; tp < 2; ++tp) {
            v8f g[4];
#pragma unroll
            for (int q = 0; q < 4; ++q) {
                const int t = tp + 2 * q;
                const int n = t * 16 + np;
                const float bias = sb[biasIdx][n];
                v8f c;
                if (xIdx >= 0) {  // fold bias + rank-1 (IN=1) term into the C operand
                    const float wv = swcol[xIdx][n];
#pragma unroll
                    for (int r = 0; r < 8; ++r) c[r] = __builtin_fmaf(xr[r], wv, bias);
                } else {
#pragma unroll
                    for (int r = 0; r < 8; ++r) c[r] = bias;
                }
                c = __builtin_amdgcn_wmma_f32_16x16x32_f16(false, A1, false, loadB(mat1, t),
                                                           (short)0, c, false, false);
                if (dual)
                    c = __builtin_amdgcn_wmma_f32_16x16x32_f16(false, A2, false, loadB(mat2, t),
                                                               (short)0, c, false, false);
                g[q] = c;
            }
#pragma unroll
            for (int r = 0; r < 8; ++r) {
                const float cprev = cst[tp * 8 + r];
                const float c2 = fsig(g[1][r]) * cprev + fsig(g[0][r]) * ftanh(g[2][r]);
                const float h  = fsig(g[3][r]) * ftanh(c2);
                cst[tp * 8 + r] = c2;
                hRow[hDst][(lane >> 4) * 8 + r][tp * 16 + np] = (_Float16)h;  // C->row-major
            }
        }
    };

    const v16h dummy = {};

#pragma unroll 1
    for (int s = 0; s < SEQ; ++s) {
        if (lane < MTILE) {
            sx[0][lane] = x[(size_t)s * BATCH + b0r + lane];
            sx[1][lane] = x[(size_t)((SEQ - s) & (SEQ - 1)) * BATCH + b0r + lane];  // (future-i) mod S, future=0
        }
        { v16h A  = loadA(0);                      cell(0,  0, A,  0, false, dummy, 0, c_f0, 0); }
        { v16h A1 = loadA(0); v16h A2 = loadA(1);  cell(1, -1, A1, 1, true,  A2,    2, c_f1, 1); }
        { v16h A  = loadA(2);                      cell(2,  1, A,  3, false, dummy, 0, c_b0, 2); }
        // replicated bug: backward layer-1 h-state is the fresh hf1
        { v16h A1 = loadA(2); v16h A2 = loadA(1);  cell(3, -1, A1, 4, true,  A2,    5, c_b1, 3); }

        // out[m, s] = [hf1|hb1][m,:] . Wlin + blin  (OUT=1)
        float acc = sblin;
        {
            const v8h* hf = (const v8h*)&hRow[1][np][0];
            const v8h* hb = (const v8h*)&hRow[3][np][0];
#pragma unroll
            for (int c4 = 0; c4 < 4; ++c4) {
                v8h f8 = hf[c4], b8 = hb[c4];
#pragma unroll
                for (int j = 0; j < 8; ++j) {
                    acc += (float)f8[j] * swlin[c4 * 8 + j] + (float)b8[j] * swlin[HID + c4 * 8 + j];
                }
            }
        }
        if (lane < MTILE) out[(size_t)(b0r + lane) * SEQ + s] = acc;
    }
}

extern "C" void kernel_launch(void* const* d_in, const int* in_sizes, int n_in,
                              void* d_out, int out_size, void* d_ws, size_t ws_size,
                              hipStream_t stream) {
    (void)in_sizes; (void)n_in; (void)out_size; (void)d_ws; (void)ws_size;
    const float* x      = (const float*)d_in[0];
    const float* wih_f0 = (const float*)d_in[1];
    const float* whh_f0 = (const float*)d_in[2];
    const float* b_f0   = (const float*)d_in[3];
    const float* wih_f1 = (const float*)d_in[4];
    const float* whh_f1 = (const float*)d_in[5];
    const float* b_f1   = (const float*)d_in[6];
    const float* wih_b0 = (const float*)d_in[7];
    const float* whh_b0 = (const float*)d_in[8];
    const float* b_b0   = (const float*)d_in[9];
    const float* wih_b1 = (const float*)d_in[10];
    const float* whh_b1 = (const float*)d_in[11];
    const float* b_b1   = (const float*)d_in[12];
    const float* wlin   = (const float*)d_in[13];
    const float* blin   = (const float*)d_in[14];
    // d_in[15] = future (always 0; backward index folded into kernel)
    float* out = (float*)d_out;

    dim3 grid(BATCH / MTILE);  // 128 independent batch tiles
    dim3 block(32);            // one wave
    bilstm_wmma_kernel<<<grid, block, 0, stream>>>(
        x, wih_f0, whh_f0, b_f0, wih_f1, whh_f1, b_f1,
        wih_b0, whh_b0, b_b0, wih_b1, whh_b1, b_b1, wlin, blin, out);
}
